// SubMConv2dASPP_25683904430829
// MI455X (gfx1250) — matrix operand order repaired and checked by reading, not compiled, not run on hardware
//
#include <hip/hip_runtime.h>
#include <hip/hip_bf16.h>

typedef __attribute__((ext_vector_type(16))) __bf16        v16bf;
typedef __attribute__((ext_vector_type(8)))  float         v8f;
typedef __attribute__((ext_vector_type(4)))  unsigned int  v4u;

#define DEVINL static __device__ __forceinline__

constexpr int Bb = 4, Hh = 512, Ww = 512;
constexpr int GRID_N = Bb * Hh * Ww;
constexpr int NUM_INS = 64;
constexpr float EPSV = 1e-5f;

DEVINL unsigned short f2bf(float f) {
    unsigned int x = __float_as_uint(f);
    unsigned int r = x + 0x7FFFu + ((x >> 16) & 1u);   // round-to-nearest-even
    return (unsigned short)(r >> 16);
}

// ---------------- setup kernels ----------------

__global__ void k_fill_i32(int* p, int n, int val) {
    int i = blockIdx.x * blockDim.x + threadIdx.x;
    if (i < n) p[i] = val;
}

__global__ void k_zero_f32(float* p, int n) {
    int i = blockIdx.x * blockDim.x + threadIdx.x;
    if (i < n) p[i] = 0.0f;
}

__global__ void k_scatter_grid(const int* __restrict__ coords, int* __restrict__ grid, int n) {
    int i = blockIdx.x * blockDim.x + threadIdx.x;
    if (i >= n) return;
    int b = coords[3 * i], y = coords[3 * i + 1], x = coords[3 * i + 2];
    grid[(b * Hh + y) * Ww + x] = i;
}

__global__ void k_build_nbr(const int* __restrict__ coords, const int* __restrict__ grid,
                            int* __restrict__ nbr, int n) {
    int i = blockIdx.x * blockDim.x + threadIdx.x;
    if (i >= n) return;
    int b = coords[3 * i], y = coords[3 * i + 1], x = coords[3 * i + 2];
    const int rates[3] = {1, 3, 5};
    #pragma unroll
    for (int r = 0; r < 3; ++r) {
        int rate = rates[r];
        #pragma unroll
        for (int t = 0; t < 9; ++t) {
            int ny = y + (t / 3 - 1) * rate;
            int nx = x + (t % 3 - 1) * rate;
            int v = -1;
            if (ny >= 0 && ny < Hh && nx >= 0 && nx < Ww)
                v = grid[(b * Hh + ny) * Ww + nx];
            nbr[(size_t)(r * 9 + t) * (size_t)n + i] = v;
        }
    }
}

__global__ void k_cvt_feats(const float* __restrict__ f, unsigned short* __restrict__ fx,
                            unsigned short* __restrict__ cat, int n) {
    int g = blockIdx.x * blockDim.x + threadIdx.x;
    if (g >= n * 64) return;
    int i = g >> 6, ch = g & 63;
    unsigned short h = f2bf(f[g]);
    fx[g] = h;
    cat[(size_t)i * 256 + ch] = h;
}

// w: [9][ck][64] f32  ->  wt: [9][64][ck] bf16 (transposed, n-major)
__global__ void k_cvt_w(const float* __restrict__ w, unsigned short* __restrict__ wt, int ck) {
    int g = blockIdx.x * blockDim.x + threadIdx.x;
    int tot = 9 * ck * 64;
    if (g >= tot) return;
    int tap = g / (ck * 64);
    int rem = g - tap * ck * 64;
    int nn = rem / ck;
    int kk = rem - nn * ck;
    wt[g] = f2bf(w[(size_t)tap * ck * 64 + (size_t)kk * 64 + nn]);
}

__global__ void k_count(const int* __restrict__ seg, float* __restrict__ cnt, int n) {
    int i = blockIdx.x * blockDim.x + threadIdx.x;
    if (i < n) atomicAdd(&cnt[seg[i]], 1.0f);
}

// ---------------- WMMA conv kernel ----------------
// Each wave computes a 16-point x 64-channel output tile with bf16 WMMA, f32 accum.

union FragBF { v16bf v; v4u q[2]; };

template <int CK>
__global__ __launch_bounds__(256) void k_conv_wmma(
    const unsigned short* __restrict__ X,    // [n][CK] bf16
    const int*            __restrict__ nbr,  // [9][n] neighbor indices (-1 = none)
    const unsigned short* __restrict__ Wt,   // [9][64][CK] bf16 (transposed)
    const float*          __restrict__ bias, // [64]
    float*                __restrict__ Y,    // [n][64]
    int n)
{
    const int lane = threadIdx.x & 31;
    const int wave = threadIdx.x >> 5;
    const int ntiles = (n + 15) >> 4;
    const int tile = blockIdx.x * 8 + wave;
    if (tile >= ntiles) return;              // wave-uniform: EXEC stays full for WMMA

    const int m0   = tile << 4;
    const int lrow = lane & 15;
    const int mrow = m0 + lrow;
    const bool rowOK = (mrow < n);
    const int hi = (lane >> 4) & 1;

    v8f acc[4];
    #pragma unroll
    for (int nb = 0; nb < 4; ++nb) {
        float bv = bias[nb * 16 + lrow];
        #pragma unroll
        for (int r = 0; r < 8; ++r) acc[nb][r] = bv;
    }

    const v4u zq = {0u, 0u, 0u, 0u};

    for (int tap = 0; tap < 9; ++tap) {
        int nidx = rowOK ? nbr[(size_t)tap * (size_t)n + mrow] : -1;
        const bool valid = (nidx >= 0);
        const unsigned short* arow = X + (size_t)(valid ? nidx : 0) * CK;

        if (tap < 8) {                       // gfx1250 global_prefetch_b8 for next tap
            int pn = rowOK ? nbr[(size_t)(tap + 1) * (size_t)n + mrow] : -1;
            __builtin_prefetch(X + (size_t)(pn < 0 ? 0 : pn) * CK, 0, 0);
        }

        const unsigned short* wtap = Wt + (size_t)tap * 64 * CK;

        #pragma unroll
        for (int ks = 0; ks < CK; ks += 32) {
            // A fragment (16x32 bf16): lanes<16 hold K=ks..+7 & ks+16..+23 for M=lrow,
            // lanes>=16 hold K=ks+8..+15 & ks+24..+31 (ISA A-layout interleave).
            FragBF a;
            v4u a0 = *(const v4u*)(arow + ks + hi * 8);
            v4u a1 = *(const v4u*)(arow + ks + 16 + hi * 8);
            a.q[0] = valid ? a0 : zq;
            a.q[1] = valid ? a1 : zq;

            #pragma unroll
            for (int nb = 0; nb < 4; ++nb) {
                // B fragment (32x16 bf16): lanes<16 hold K=ks..+7 & ks+16..+23 of column n,
                // lanes>=16 hold K=ks+8..+15 & ks+24..+31 (same interleave as A).
                const unsigned short* wrow = wtap + (size_t)(nb * 16 + lrow) * CK + ks + hi * 8;
                FragBF b;
                b.q[0] = *(const v4u*)(wrow);
                b.q[1] = *(const v4u*)(wrow + 16);
                acc[nb] = __builtin_amdgcn_wmma_f32_16x16x32_bf16(
                    false, a.v, false, b.v, (short)0, acc[nb], false, false);
            }
        }
    }

    // D layout: VGPR r, lanes0-15 -> M=r, lanes16-31 -> M=8+r; N = lane&15.
    #pragma unroll
    for (int nb = 0; nb < 4; ++nb) {
        #pragma unroll
        for (int r = 0; r < 8; ++r) {
            int mm = m0 + hi * 8 + r;
            if (mm < n) Y[(size_t)mm * 64 + nb * 16 + lrow] = acc[nb][r];
        }
    }
}

// ---------------- instance norm ----------------

__global__ __launch_bounds__(256) void k_stats(const float* __restrict__ Y,
                                               const int* __restrict__ seg,
                                               float* __restrict__ gsum, float* __restrict__ gss,
                                               int n, int ptsPerBlk) {
    __shared__ float ls[NUM_INS * 64];
    __shared__ float lq[NUM_INS * 64];
    for (int j = threadIdx.x; j < NUM_INS * 64; j += 256) { ls[j] = 0.f; lq[j] = 0.f; }
    __syncthreads();

    int ch  = threadIdx.x & 63;
    int sub = threadIdx.x >> 6;              // 4 point-lanes x 64 channels
    int start = blockIdx.x * ptsPerBlk;
    int end = start + ptsPerBlk; if (end > n) end = n;
    for (int i = start + sub; i < end; i += 4) {
        int s = seg[i];
        float v = Y[(size_t)i * 64 + ch];
        atomicAdd(&ls[s * 64 + ch], v);
        atomicAdd(&lq[s * 64 + ch], v * v);
    }
    __syncthreads();
    for (int j = threadIdx.x; j < NUM_INS * 64; j += 256) {
        float a = ls[j], b = lq[j];
        if (a != 0.f) atomicAdd(&gsum[j], a);
        if (b != 0.f) atomicAdd(&gss[j], b);
    }
}

__global__ void k_moments(const float* __restrict__ gsum, const float* __restrict__ gss,
                          const float* __restrict__ cnt,
                          float* __restrict__ mean, float* __restrict__ inv) {
    int j = blockIdx.x * blockDim.x + threadIdx.x;
    if (j >= NUM_INS * 64) return;
    float c = fmaxf(cnt[j >> 6], 1.0f);
    float m = gsum[j] / c;
    float v = gss[j] / c - m * m;
    mean[j] = m;
    inv[j]  = rsqrtf(v + EPSV);
}

__global__ void k_norm_relu(const float* __restrict__ Y, const int* __restrict__ seg,
                            const float* __restrict__ mean, const float* __restrict__ inv,
                            unsigned short* __restrict__ cat, int colOff, int n) {
    int g = blockIdx.x * blockDim.x + threadIdx.x;
    if (g >= n * 64) return;
    int i = g >> 6, ch = g & 63;
    int s = seg[i];
    float v = (Y[g] - mean[s * 64 + ch]) * inv[s * 64 + ch];
    v = fmaxf(v, 0.0f);
    cat[(size_t)i * 256 + colOff + ch] = f2bf(v);
}

// ---------------- launch ----------------

extern "C" void kernel_launch(void* const* d_in, const int* in_sizes, int n_in,
                              void* d_out, int out_size, void* d_ws, size_t ws_size,
                              hipStream_t stream) {
    (void)n_in; (void)out_size; (void)ws_size;

    const float* features = (const float*)d_in[0];
    const int*   coords   = (const int*)d_in[1];
    const int*   seg      = (const int*)d_in[2];
    // d_in[3] = num_instances scalar (== 64, compiled in)
    const float* w1 = (const float*)d_in[4];  const float* b1 = (const float*)d_in[5];
    const float* w2 = (const float*)d_in[6];  const float* b2 = (const float*)d_in[7];
    const float* w3 = (const float*)d_in[8];  const float* b3 = (const float*)d_in[9];
    const float* wf = (const float*)d_in[10]; const float* bf = (const float*)d_in[11];

    const int N = in_sizes[0] / 64;

    char* p = (char*)d_ws;
    auto alloc = [&](size_t bytes) -> char* {
        char* r = p; p += (bytes + 255) & ~(size_t)255; return r;
    };
    int*            grid = (int*)           alloc((size_t)GRID_N * 4);
    int*            nbr  = (int*)           alloc((size_t)27 * N * 4);
    unsigned short* fx   = (unsigned short*)alloc((size_t)N * 64 * 2);
    unsigned short* cat  = (unsigned short*)alloc((size_t)N * 256 * 2);
    float*          ybuf = (float*)         alloc((size_t)N * 64 * 4);
    unsigned short* wT   = (unsigned short*)alloc((size_t)3 * 9 * 64 * 64 * 2);
    unsigned short* wfT  = (unsigned short*)alloc((size_t)9 * 64 * 256 * 2);
    float*          cnt  = (float*)         alloc(64 * 4);
    float*          sum  = (float*)         alloc(4096 * 4);
    float*          ss   = (float*)         alloc(4096 * 4);
    float*          mean = (float*)         alloc(4096 * 4);
    float*          inv  = (float*)         alloc(4096 * 4);

    constexpr int TB = 256;

    k_fill_i32    <<<(GRID_N + TB - 1) / TB, TB, 0, stream>>>(grid, GRID_N, -1);
    k_scatter_grid<<<(N + TB - 1) / TB,      TB, 0, stream>>>(coords, grid, N);
    k_build_nbr   <<<(N + TB - 1) / TB,      TB, 0, stream>>>(coords, grid, nbr, N);
    k_cvt_feats   <<<(N * 64 + TB - 1) / TB, TB, 0, stream>>>(features, fx, cat, N);

    k_cvt_w<<<(9 * 64 * 64 + TB - 1) / TB,  TB, 0, stream>>>(w1, wT + (size_t)0 * 9 * 4096, 64);
    k_cvt_w<<<(9 * 64 * 64 + TB - 1) / TB,  TB, 0, stream>>>(w2, wT + (size_t)1 * 9 * 4096, 64);
    k_cvt_w<<<(9 * 64 * 64 + TB - 1) / TB,  TB, 0, stream>>>(w3, wT + (size_t)2 * 9 * 4096, 64);
    k_cvt_w<<<(9 * 256 * 64 + TB - 1) / TB, TB, 0, stream>>>(wf, wfT, 256);

    k_zero_f32<<<1, 64, 0, stream>>>(cnt, 64);
    k_count   <<<(N + TB - 1) / TB, TB, 0, stream>>>(seg, cnt, N);

    const int ntiles = (N + 15) / 16;
    const int nblk   = (ntiles + 7) / 8;
    const int ptsPerBlk = 1024;
    const int statBlks  = (N + ptsPerBlk - 1) / ptsPerBlk;
    const float* biases[3] = {b1, b2, b3};

    for (int br = 0; br < 3; ++br) {
        k_zero_f32<<<(4096 + TB - 1) / TB, TB, 0, stream>>>(sum, 4096);
        k_zero_f32<<<(4096 + TB - 1) / TB, TB, 0, stream>>>(ss, 4096);
        k_conv_wmma<64><<<nblk, TB, 0, stream>>>(
            fx, nbr + (size_t)br * 9 * N, wT + (size_t)br * 9 * 4096, biases[br], ybuf, N);
        k_stats  <<<statBlks, TB, 0, stream>>>(ybuf, seg, sum, ss, N, ptsPerBlk);
        k_moments<<<(4096 + TB - 1) / TB, TB, 0, stream>>>(sum, ss, cnt, mean, inv);
        k_norm_relu<<<(N * 64 + TB - 1) / TB, TB, 0, stream>>>(
            ybuf, seg, mean, inv, cat, 64 + br * 64, N);
    }

    // Final 3x3 conv (rate 1) over concat[.,256] -> d_out f32 [N][64]
    k_conv_wmma<256><<<nblk, TB, 0, stream>>>(cat, nbr, wfT, bf, (float*)d_out, N);
}
